// SuperfeatureLoss_7696581394670
// MI455X (gfx1250) — compile-verified, matrix operand order checked by reading, not run on hardware
//
#include <hip/hip_runtime.h>
#include <hip/hip_bf16.h>

// ---------------------------------------------------------------------------
// SuperfeatureLoss on MI455X (gfx1250, wave32, WMMA)
//
// Heavy part: S = qn @ pn^T  (4096 x 4096 x 512, 17.2 GFLOP) done with
// v_wmma_f32_16x16x32_f16, with the argmin/argmax reductions fused into the
// GEMM epilogue so the 64MB distance matrix never exists in memory.
// qh/ph (4MB each, f16) are fully L2-resident (192MB L2), so the 256x reuse
// of q across column-stripe workgroups is L2 traffic, not HBM.
// The per-WG p-stripe is staged into LDS with the CDNA5 async copy path
// (global_load_async_to_lds_b128 + s_wait_asynccnt), no VGPR round-trip.
// ---------------------------------------------------------------------------

#define NFEAT 4096
#define DFEAT 512
#define NNEG  5

static constexpr float kMargin = 1.1f;
static constexpr float kEps    = 1e-6f;
static constexpr float kLowe   = 0.9f;
static constexpr float kWeight = 1.0f;

typedef __attribute__((ext_vector_type(16))) _Float16 v16h;
typedef __attribute__((ext_vector_type(8)))  _Float16 v8h;
typedef __attribute__((ext_vector_type(4)))  _Float16 v4h;
typedef __attribute__((ext_vector_type(8)))  float    v8f;

__device__ __forceinline__ float waveSum(float v) {
#pragma unroll
  for (int m = 16; m > 0; m >>= 1) v += __shfl_xor(v, m, 32);
  return v;
}

// Order-preserving float->uint, packed with ~idx so that larger value wins and
// ties prefer the SMALLER index (jnp.argmin/argmax first-occurrence rule).
__device__ __forceinline__ unsigned long long packKey(float v, int idx) {
  unsigned u = __float_as_uint(v);
  u = (u & 0x80000000u) ? ~u : (u | 0x80000000u);
  return ((unsigned long long)u << 32) | (unsigned)(~(unsigned)idx);
}

// ---------------------------------------------------------------------------
// Kernel 1: one wave32 per feature index i.
// Normalizes q,p,negs; writes f16 qn/pn; computes per_i (contrastive terms),
// diagonal distance dist[i][i], 0.5*||qn||^2, 0.5*||pn||^2; zeroes rowBest.
// ---------------------------------------------------------------------------
__global__ void __launch_bounds__(256)
k_prep(const float* __restrict__ feats,
       _Float16* __restrict__ qh, _Float16* __restrict__ ph,
       float* __restrict__ perI, float* __restrict__ dDiag,
       float* __restrict__ hA2, float* __restrict__ hB2,
       unsigned long long* __restrict__ rowBest)
{
  const int lane = threadIdx.x & 31;
  const int row  = blockIdx.x * (blockDim.x >> 5) + (threadIdx.x >> 5);
  if (row >= NFEAT) return;

  const float* q = feats + (size_t)row * DFEAT;
  const float* p = feats + (size_t)NFEAT * DFEAT + (size_t)row * DFEAT;

  float4 qv[4], pv[4];
#pragma unroll
  for (int s = 0; s < 4; ++s) {
    qv[s] = ((const float4*)q)[lane + 32 * s];
    pv[s] = ((const float4*)p)[lane + 32 * s];
  }
  float* qf = (float*)qv;
  float* pf = (float*)pv;

  float sq = 0.f, sp = 0.f;
#pragma unroll
  for (int e = 0; e < 16; ++e) { sq += qf[e] * qf[e]; sp += pf[e] * pf[e]; }
  const float invq = 1.f / fmaxf(sqrtf(waveSum(sq)), 1e-12f);
  const float invp = 1.f / fmaxf(sqrtf(waveSum(sp)), 1e-12f);
#pragma unroll
  for (int e = 0; e < 16; ++e) { qf[e] *= invq; pf[e] *= invp; }

  // a2, b2, diag dot, d_pos on the normalized vectors (mirrors reference)
  float a2 = 0.f, b2 = 0.f, sd = 0.f, dp = 0.f;
#pragma unroll
  for (int e = 0; e < 16; ++e) {
    a2 += qf[e] * qf[e];
    b2 += pf[e] * pf[e];
    sd += qf[e] * pf[e];
    const float d = qf[e] - pf[e] + kEps;
    dp += d * d;
  }
  a2 = waveSum(a2); b2 = waveSum(b2); sd = waveSum(sd); dp = waveSum(dp);
  const float dpos = sqrtf(dp);
  float per = 0.5f * dpos * dpos;
  const float ddg = sqrtf(fmaxf(a2 + b2 - 2.f * sd, 0.f));

  // store f16 normalized rows (row-major, 8-byte packed stores)
  v4h* qdst = (v4h*)(qh + (size_t)row * DFEAT);
  v4h* pdst = (v4h*)(ph + (size_t)row * DFEAT);
#pragma unroll
  for (int s = 0; s < 4; ++s) {
    v4h tq, tp;
#pragma unroll
    for (int e = 0; e < 4; ++e) {
      tq[e] = (_Float16)qf[4 * s + e];
      tp[e] = (_Float16)pf[4 * s + e];
    }
    qdst[lane + 32 * s] = tq;
    pdst[lane + 32 * s] = tp;
  }

  // negatives: contrastive hinge terms
#pragma unroll 1
  for (int n = 0; n < NNEG; ++n) {
    const float* g = feats + (size_t)(2 + n) * NFEAT * DFEAT + (size_t)row * DFEAT;
    float4 gv[4];
#pragma unroll
    for (int s = 0; s < 4; ++s) gv[s] = ((const float4*)g)[lane + 32 * s];
    float* gf = (float*)gv;
    float sg = 0.f;
#pragma unroll
    for (int e = 0; e < 16; ++e) sg += gf[e] * gf[e];
    const float invg = 1.f / fmaxf(sqrtf(waveSum(sg)), 1e-12f);
    float dn2 = 0.f;
#pragma unroll
    for (int e = 0; e < 16; ++e) {
      const float d = qf[e] - gf[e] * invg + kEps;
      dn2 += d * d;
    }
    const float dn = sqrtf(waveSum(dn2));
    const float h = fmaxf(kMargin - dn, 0.f);
    per += 0.5f * h * h;
  }

  if (lane == 0) {
    perI[row]  = per;
    dDiag[row] = ddg;
    hA2[row]   = 0.5f * a2;
    hB2[row]   = 0.5f * b2;
    rowBest[row] = 0ull;
  }
}

// ---------------------------------------------------------------------------
// Kernel 2: fused GEMM + argmax reductions.
// Each workgroup owns 16 columns (p rows j0..j0+15, async-staged into LDS;
// the compiler then promotes the full B stripe into registers).
// 4 waves x 4 row-subtiles cover 256 rows per iteration (16 iterations).
// Row argmin(dist) -> packed atomicMax on key (s_ij - 0.5*b2_j).
// Column top-2 argmin(dist) kept per-lane on key (s_ij - 0.5*a2_i), merged
// via LDS at the end (workgroup-local: no inter-block merge needed).
// ---------------------------------------------------------------------------
__global__ void __launch_bounds__(128)
k_gemm(const _Float16* __restrict__ qh, const _Float16* __restrict__ ph,
       const float* __restrict__ hA2, const float* __restrict__ hB2,
       unsigned long long* __restrict__ rowBest,
       int* __restrict__ top1, int* __restrict__ top2)
{
  __shared__ __align__(16) _Float16 plds[16 * DFEAT];
  __shared__ float msV1[8][16], msV2[8][16];
  __shared__ int   msI1[8][16], msI2[8][16];

  const int tid  = threadIdx.x;
  const int lane = tid & 31;
  const int w    = tid >> 5;
  const int j0   = blockIdx.x * 16;

  // ---- async-stage the contiguous 16KB p-stripe into LDS ------------------
  // CDNA5 path: memory -> LDS without a VGPR round-trip, tracked by ASYNCcnt.
  {
    const char* gsrc = (const char*)(ph + (size_t)j0 * DFEAT);
    const unsigned lbase =
        (unsigned)(unsigned long long)(uintptr_t)(void*)&plds[0];
#pragma unroll
    for (int t = 0; t < 8; ++t) {
      const unsigned laddr = lbase + (unsigned)(tid * 16 + t * 2048);
      const char* ga = gsrc + (size_t)(tid * 16 + t * 2048);
      asm volatile("global_load_async_to_lds_b128 %0, %1, off"
                   :
                   : "v"(laddr), "v"(ga)
                   : "memory");
    }
    asm volatile("s_wait_asynccnt 0x0" ::: "memory");
  }
  __syncthreads();

  const int half = lane >> 4;       // 0: K 0..7/16..23, 1: K 8..15/24..31
  const int l16  = lane & 15;       // A: row-in-tile, B/C: col-in-tile
  const int col  = j0 + l16;
  const float hb2c = hB2[col];      // 0.5*||pn_col||^2
  const int kb = half * 8;

  float cV1 = -__builtin_inff(), cV2 = -__builtin_inff();
  int   cI1 = 0x7fffffff,        cI2 = 0x7fffffff;

  const _Float16* bbase = plds + (size_t)l16 * DFEAT + kb;

  for (int it = 0; it < NFEAT / 256; ++it) {
    const int rbase = it * 256 + w * 64;

    // prefetch next iteration's A rows into cache hierarchy
    if (it + 1 < NFEAT / 256)
      __builtin_prefetch(qh + (size_t)(rbase + 256 + l16) * DFEAT, 0, 1);

    v8f c[4] = {};
#pragma unroll
    for (int kc = 0; kc < DFEAT; kc += 32) {
      // B fragment (shared by 4 row-subtiles): 32x16, lanes = N
      v8h blo = *(const v8h*)(bbase + kc);
      v8h bhi = *(const v8h*)(bbase + kc + 16);
      v16h b;
#pragma unroll
      for (int e = 0; e < 8; ++e) { b[e] = blo[e]; b[8 + e] = bhi[e]; }
#pragma unroll
      for (int sub = 0; sub < 4; ++sub) {
        const _Float16* aptr =
            qh + (size_t)(rbase + sub * 16 + l16) * DFEAT + kb;
        v8h alo = *(const v8h*)(aptr + kc);
        v8h ahi = *(const v8h*)(aptr + kc + 16);
        v16h a;
#pragma unroll
        for (int e = 0; e < 8; ++e) { a[e] = alo[e]; a[8 + e] = ahi[e]; }
        c[sub] = __builtin_amdgcn_wmma_f32_16x16x32_f16(
            false, a, false, b, (short)0, c[sub], false, false);
      }
    }

    // epilogue reductions. C layout: element v of lane l =
    //   S[r0 + v + 8*(l>>4)][j0 + (l&15)]
#pragma unroll
    for (int sub = 0; sub < 4; ++sub) {
      const int r0 = rbase + sub * 16;
#pragma unroll
      for (int v = 0; v < 8; ++v) {
        const int r = r0 + v + 8 * half;
        const float sraw = c[sub][v];

        // per-row argmin(dist) over this 16-col stripe: key s - 0.5*b2_j
        float bv = sraw - hb2c;
        int   bj = col;
#pragma unroll
        for (int m = 1; m < 16; m <<= 1) {
          const float ov = __shfl_xor(bv, m, 32);
          const int   oj = __shfl_xor(bj, m, 32);
          if (ov > bv || (ov == bv && oj < bj)) { bv = ov; bj = oj; }
        }
        if (l16 == 0) atomicMax(&rowBest[r], packKey(bv, bj));

        // per-column top-2 over rows: key s - 0.5*a2_i
        const float ck = sraw - hA2[r];
        if (ck > cV1 || (ck == cV1 && r < cI1)) {
          cV2 = cV1; cI2 = cI1; cV1 = ck; cI1 = r;
        } else if (ck > cV2 || (ck == cV2 && r < cI2)) {
          cV2 = ck; cI2 = r;
        }
      }
    }
  }

  // merge 8 partial top-2 lists (4 waves x 2 lane-halves) per column
  const int slot = w * 2 + half;
  msV1[slot][l16] = cV1; msI1[slot][l16] = cI1;
  msV2[slot][l16] = cV2; msI2[slot][l16] = cI2;
  __syncthreads();

  if (tid < 16) {
    float V1 = -__builtin_inff(), V2 = -__builtin_inff();
    int   I1 = 0x7fffffff,        I2 = 0x7fffffff;
#pragma unroll
    for (int s = 0; s < 8; ++s) {
#pragma unroll
      for (int e = 0; e < 2; ++e) {
        const float vv = e ? msV2[s][tid] : msV1[s][tid];
        const int   ii = e ? msI2[s][tid] : msI1[s][tid];
        if (vv > V1 || (vv == V1 && ii < I1)) {
          V2 = V1; I2 = I1; V1 = vv; I1 = ii;
        } else if (vv > V2 || (vv == V2 && ii < I2)) {
          V2 = vv; I2 = ii;
        }
      }
    }
    top1[j0 + tid] = I1;
    top2[j0 + tid] = I2;
  }
}

// ---------------------------------------------------------------------------
// Kernel 3: mask + masked sum.
// mask[j] = (best2[j]==j) && (best1[j]==j) && (dist[j][j]/(float)second_idx[j] <= 0.9)
// (ratio divides by the INDEX, faithfully matching the reference.)
// ---------------------------------------------------------------------------
__global__ void __launch_bounds__(256)
k_final(const unsigned long long* __restrict__ rowBest,
        const int* __restrict__ top1, const int* __restrict__ top2,
        const float* __restrict__ dDiag, const float* __restrict__ perI,
        float* __restrict__ out)
{
  __shared__ float red[256];
  float acc = 0.f;
  for (int j = threadIdx.x; j < NFEAT; j += 256) {
    const unsigned long long rb = rowBest[j];
    const int b1 = (int)(~(unsigned)rb);        // row j's argmin column
    bool m = (top1[j] == j) && (b1 == j);
    const float ratio = dDiag[j] / (float)top2[j];
    m = m && (ratio <= kLowe);
    if (m) acc += perI[j];
  }
  red[threadIdx.x] = acc;
  __syncthreads();
  for (int s = 128; s > 0; s >>= 1) {
    if (threadIdx.x < s) red[threadIdx.x] += red[threadIdx.x + s];
    __syncthreads();
  }
  if (threadIdx.x == 0) out[0] = red[0] * kWeight;
}

// ---------------------------------------------------------------------------
extern "C" void kernel_launch(void* const* d_in, const int* in_sizes, int n_in,
                              void* d_out, int out_size, void* d_ws, size_t ws_size,
                              hipStream_t stream)
{
  (void)in_sizes; (void)n_in; (void)out_size; (void)ws_size;
  const float* feats = (const float*)d_in[0];   // [7, 4096, 512] f32
  // d_in[1] (target) is structural only; unused by the loss value.

  char* ws = (char*)d_ws;
  size_t off = 0;
  _Float16* qh = (_Float16*)(ws + off); off += (size_t)NFEAT * DFEAT * 2;  // 4MB
  _Float16* ph = (_Float16*)(ws + off); off += (size_t)NFEAT * DFEAT * 2;  // 4MB
  float* perI  = (float*)(ws + off); off += (size_t)NFEAT * 4;
  float* dDiag = (float*)(ws + off); off += (size_t)NFEAT * 4;
  float* hA2   = (float*)(ws + off); off += (size_t)NFEAT * 4;
  float* hB2   = (float*)(ws + off); off += (size_t)NFEAT * 4;
  unsigned long long* rowBest = (unsigned long long*)(ws + off);
  off += (size_t)NFEAT * 8;
  int* top1 = (int*)(ws + off); off += (size_t)NFEAT * 4;
  int* top2 = (int*)(ws + off); off += (size_t)NFEAT * 4;

  hipLaunchKernelGGL(k_prep, dim3(NFEAT / 8), dim3(256), 0, stream,
                     feats, qh, ph, perI, dDiag, hA2, hB2, rowBest);
  hipLaunchKernelGGL(k_gemm, dim3(NFEAT / 16), dim3(128), 0, stream,
                     qh, ph, hA2, hB2, rowBest, top1, top2);
  hipLaunchKernelGGL(k_final, dim3(1), dim3(256), 0, stream,
                     rowBest, top1, top2, dDiag, perI, (float*)d_out);
}
